// AttentionLayerBlock_2662879723833
// MI455X (gfx1250) — matrix operand order repaired and checked by reading, not compiled, not run on hardware
//
#include <hip/hip_runtime.h>
#include <math.h>

typedef __attribute__((ext_vector_type(8)))  float   v8f;
typedef __attribute__((ext_vector_type(16))) __bf16  v16bf;
typedef __attribute__((ext_vector_type(8)))  __bf16  v8bf;

#define B_      4
#define C_      192
#define HW_     16384
#define NLOC_   65536
#define QKVC_   576
#define HID_    384
#define PINC_   768

// ---------------------------------------------------------------- f32 -> bf16
__global__ void cvt_bf16_kernel(const float* __restrict__ src, __bf16* __restrict__ dst, int n) {
    int i = blockIdx.x * 256 + threadIdx.x;
    if (i < n) dst[i] = (__bf16)src[i];
}

// ------------------------------------------------- LayerNorm over C, NCHW -> [n][C] bf16
__global__ void ln_kernel(const float* __restrict__ x, const float* __restrict__ w,
                          const float* __restrict__ bia, __bf16* __restrict__ out) {
    __shared__ __bf16 lds[128 * C_];
    const int tid = threadIdx.x;                 // 128 threads, one location each
    const int n   = blockIdx.x * 128 + tid;
    const int b   = n >> 14;
    const int s   = n & (HW_ - 1);
    const float* xp = x + ((size_t)b * C_) * HW_ + s;
    float sum = 0.f, sq = 0.f;
    for (int c = 0; c < C_; ++c) {
        float v = xp[(size_t)c * HW_];
        sum += v; sq += v * v;
    }
    const float mu   = sum * (1.0f / C_);
    const float var  = sq * (1.0f / C_) - mu * mu;
    const float rstd = rsqrtf(var + 1e-5f);
    for (int c = 0; c < C_; ++c) {
        float v = xp[(size_t)c * HW_];
        lds[tid * C_ + c] = (__bf16)((v - mu) * rstd * w[c] + bia[c]);
    }
    __syncthreads();
    // coalesced copy-out ([n][C] bf16 block is contiguous)
    unsigned* dstu = (unsigned*)(out + (size_t)(blockIdx.x * 128) * C_);
    const unsigned* ldsu = (const unsigned*)lds;
    for (int i = tid; i < 128 * C_ / 2; i += 128) dstu[i] = ldsu[i];
}

// ------------------------------------------------- bf16 WMMA GEMM: out[m][n] = W[m][:]·Xt[n][:]
// Xt: [NLOC][K] bf16, Wb: [M][K] bf16. Weights staged in LDS via async load-to-LDS.
// Output NCHW f32 (opt residual) or NCHW bf16. Per wave: 32n x 64m tile (8 WMMA/k-step).
__global__ void gemm_kernel(const __bf16* __restrict__ Xt, const __bf16* __restrict__ Wb,
                            int K, int M, const float* __restrict__ resid,
                            float* __restrict__ outF, __bf16* __restrict__ outH) {
    __shared__ __align__(16) __bf16 wlds[64 * (384 + 8)];   // rows padded to K+8
    const int tid = threadIdx.x;
    const int m_base = blockIdx.y * 64;
    const int RS = K + 8;

    // ---- stage 64 weight rows into LDS (async gather, tracked by ASYNCcnt) ----
    {
        const int cpr   = K / 8;            // 16B chunks per row
        const int total = 64 * cpr;         // 1536 (K=192) / 3072 (K=384): exact multiples of 256
        for (int ci = tid; ci < total; ci += 256) {
            const int row = ci / cpr;
            const int col = (ci - row * cpr) * 8;
            unsigned laddr = (unsigned)(size_t)&wlds[row * RS + col];
            const __bf16* gaddr = Wb + (size_t)(m_base + row) * K + col;
            asm volatile("global_load_async_to_lds_b128 %0, %1, off"
                         :: "v"(laddr), "v"(gaddr) : "memory");
        }
        asm volatile("s_wait_asynccnt 0" ::: "memory");
        __syncthreads();
    }

    const int wave = tid >> 5;
    const int l    = tid & 31;
    const int g    = l >> 4;       // half-wave group
    const int nl   = l & 15;
    const int n_base = blockIdx.x * 256 + wave * 32;

    v8f acc[2][4] = {{v8f{0}, v8f{0}, v8f{0}, v8f{0}}, {v8f{0}, v8f{0}, v8f{0}, v8f{0}}};
    const __bf16* xr0 = Xt + (size_t)(n_base + nl) * K;
    const __bf16* xr1 = Xt + (size_t)(n_base + 16 + nl) * K;

    for (int kk = 0; kk < K; kk += 32) {
        v16bf a0, a1;
        ((v8bf*)&a0)[0] = *(const v8bf*)(xr0 + kk + 8 * g);        // K = kk+8g .. +7
        ((v8bf*)&a0)[1] = *(const v8bf*)(xr0 + kk + 16 + 8 * g);   // K = kk+16+8g .. +7
        ((v8bf*)&a1)[0] = *(const v8bf*)(xr1 + kk + 8 * g);
        ((v8bf*)&a1)[1] = *(const v8bf*)(xr1 + kk + 16 + 8 * g);
#pragma unroll
        for (int mi = 0; mi < 4; ++mi) {
            const __bf16* wl = &wlds[(mi * 16 + nl) * RS + kk + 8 * g];  // K = kk+8g .. +7, kk+16+8g .. +7
            v16bf bv;
            ((v8bf*)&bv)[0] = *(const v8bf*)(wl);
            ((v8bf*)&bv)[1] = *(const v8bf*)(wl + 16);
            acc[0][mi] = __builtin_amdgcn_wmma_f32_16x16x32_bf16(
                false, a0, false, bv, (short)0, acc[0][mi], false, false);
            acc[1][mi] = __builtin_amdgcn_wmma_f32_16x16x32_bf16(
                false, a1, false, bv, (short)0, acc[1][mi], false, false);
        }
    }

#pragma unroll
    for (int ni = 0; ni < 2; ++ni) {
        const int n0 = n_base + 16 * ni;
        const int bb = n0 >> 14;
        const int s0 = (n0 & (HW_ - 1)) + 8 * g;                 // D lane r -> spatial s0+r
#pragma unroll
        for (int mi = 0; mi < 4; ++mi) {
            const int m = m_base + mi * 16 + nl;
            const size_t base = ((size_t)bb * M + m) * HW_ + s0;
            if (outF) {
#pragma unroll
                for (int r = 0; r < 8; ++r) {
                    float v = acc[ni][mi][r];
                    if (resid) v += resid[base + r];
                    outF[base + r] = v;
                }
            } else {
#pragma unroll
                for (int r = 0; r < 8; ++r) outH[base + r] = (__bf16)acc[ni][mi][r];
            }
        }
    }
}

// ------------------------------------------------- depthwise 3x3 (f32 in/out), SAME
__global__ void dwconv_kernel(const float* __restrict__ in, const float* __restrict__ w9,
                              float* __restrict__ out, int Ctot) {
    const size_t idx = (size_t)blockIdx.x * 256 + threadIdx.x;
    const int s  = idx & (HW_ - 1);
    const int c  = (idx >> 14) % Ctot;
    const int b  = idx / ((size_t)Ctot * HW_);
    const int h  = s >> 7, wx = s & 127;
    const float* ip = in + ((size_t)b * Ctot + c) * HW_;
    const float* kp = w9 + c * 9;
    float acc = 0.f;
#pragma unroll
    for (int ky = 0; ky < 3; ++ky) {
        const int ih = h + ky - 1;
        if (ih < 0 || ih > 127) continue;
#pragma unroll
        for (int kx = 0; kx < 3; ++kx) {
            const int iw = wx + kx - 1;
            if (iw < 0 || iw > 127) continue;
            acc += ip[ih * 128 + iw] * kp[ky * 3 + kx];
        }
    }
    out[idx] = acc;
}

// ------------------------------------------------- q/k row L2-norm scales
__global__ void rownorm_kernel(const float* __restrict__ qkvd, float* __restrict__ scale) {
    __shared__ float red[256];
    const int r  = blockIdx.x;         // 4*384 rows (q: cc<192, k: 192<=cc<384)
    const int b  = r / 384, cc = r % 384;
    const float* p = qkvd + ((size_t)b * QKVC_ + cc) * HW_;
    float s = 0.f;
    for (int i = threadIdx.x; i < HW_; i += 256) { float v = p[i]; s += v * v; }
    red[threadIdx.x] = s; __syncthreads();
    for (int o = 128; o > 0; o >>= 1) {
        if (threadIdx.x < o) red[threadIdx.x] += red[threadIdx.x + o];
        __syncthreads();
    }
    if (threadIdx.x == 0) scale[r] = 1.0f / fmaxf(sqrtf(red[0]), 1e-12f);
}

// ------------------------------------------------- attn = relu(temp * q_n · k_n^T), per (b,head)
#define ACH 256
#define ACHP (ACH + 4)
__global__ void attn_kernel(const float* __restrict__ qkvd, const float* __restrict__ scale,
                            const float* __restrict__ temp, float* __restrict__ attnS) {
    __shared__ float lq[32 * ACHP];
    __shared__ float lk[32 * ACHP];
    const int bh = blockIdx.x, b = bh / 6, h = bh % 6, tid = threadIdx.x;
    const float* qb = qkvd + ((size_t)b * QKVC_ + h * 32) * HW_;
    const float* kb = qkvd + ((size_t)b * QKVC_ + C_ + h * 32) * HW_;
    float acc[4] = {0.f, 0.f, 0.f, 0.f};
    for (int n0 = 0; n0 < HW_; n0 += ACH) {
        __syncthreads();
        for (int i = tid; i < 32 * ACH; i += 256) {
            const int c = i / ACH, j = i % ACH;
            lq[c * ACHP + j] = qb[(size_t)c * HW_ + n0 + j];
            lk[c * ACHP + j] = kb[(size_t)c * HW_ + n0 + j];
        }
        __syncthreads();
#pragma unroll
        for (int pi = 0; pi < 4; ++pi) {
            const int p = tid + 256 * pi, c = p >> 5, d = p & 31;
            float s = acc[pi];
            for (int j = 0; j < ACH; ++j) s += lq[c * ACHP + j] * lk[d * ACHP + j];
            acc[pi] = s;
        }
    }
#pragma unroll
    for (int pi = 0; pi < 4; ++pi) {
        const int p = tid + 256 * pi, c = p >> 5, d = p & 31;
        float v = acc[pi] * scale[b * 384 + h * 32 + c] * scale[b * 384 + 192 + h * 32 + d] * temp[h];
        attnS[(size_t)bh * 1024 + p] = fmaxf(v, 0.f);
    }
}

// ------------------------------------------------- out = attn @ v  ->  [n][192] bf16
__global__ void attnv_kernel(const float* __restrict__ qkvd, const float* __restrict__ attnS,
                             __bf16* __restrict__ outt) {
    __shared__ float S[1024];
    const int bh = blockIdx.x >> 6, sc = blockIdx.x & 63;
    const int b = bh / 6, h = bh % 6;
    const int s = sc * 256 + threadIdx.x;
    for (int i = threadIdx.x; i < 1024; i += 256) S[i] = attnS[(size_t)bh * 1024 + i];
    __syncthreads();
    const float* vb = qkvd + ((size_t)b * QKVC_ + 2 * C_ + h * 32) * HW_ + s;
    float acc[32];
#pragma unroll
    for (int c = 0; c < 32; ++c) acc[c] = 0.f;
    for (int d = 0; d < 32; ++d) {
        const float vv = vb[(size_t)d * HW_];
#pragma unroll
        for (int c = 0; c < 32; ++c) acc[c] += S[c * 32 + d] * vv;
    }
    __bf16* op = outt + ((size_t)(b * HW_ + s)) * C_ + h * 32;
#pragma unroll
    for (int c = 0; c < 32; c += 2) {
        union { struct { __bf16 lo, hi; } p; unsigned u; } cv;
        cv.p.lo = (__bf16)acc[c]; cv.p.hi = (__bf16)acc[c + 1];
        *(unsigned*)(op + c) = cv.u;
    }
}

// ------------------------------------------------- FFN depthwise 3x3 (bf16 in) + gelu gate -> [n][384] bf16
__global__ void dwffn_gelu_kernel(const __bf16* __restrict__ in, const float* __restrict__ w9,
                                  __bf16* __restrict__ yt) {
    const size_t idx = (size_t)blockIdx.x * 256 + threadIdx.x;   // over 4*384*HW
    const int s = idx & (HW_ - 1);
    const int c = (idx >> 14) % HID_;
    const int b = idx / ((size_t)HID_ * HW_);
    const int h = s >> 7, wx = s & 127;
    float z[2] = {0.f, 0.f};
#pragma unroll
    for (int half = 0; half < 2; ++half) {
        const int ch = c + half * HID_;
        const __bf16* ip = in + ((size_t)b * PINC_ + ch) * HW_;
        const float*  kp = w9 + ch * 9;
        float acc = 0.f;
#pragma unroll
        for (int ky = 0; ky < 3; ++ky) {
            const int ih = h + ky - 1;
            if (ih < 0 || ih > 127) continue;
#pragma unroll
            for (int kx = 0; kx < 3; ++kx) {
                const int iw = wx + kx - 1;
                if (iw < 0 || iw > 127) continue;
                acc += (float)ip[ih * 128 + iw] * kp[ky * 3 + kx];
            }
        }
        z[half] = acc;
    }
    const float g = 0.5f * z[0] * (1.0f + erff(z[0] * 0.70710678118654752f));
    yt[((size_t)(b * HW_ + s)) * HID_ + c] = (__bf16)(g * z[1]);
}

// =================================================================== launch
extern "C" void kernel_launch(void* const* d_in, const int* in_sizes, int n_in,
                              void* d_out, int out_size, void* d_ws, size_t ws_size,
                              hipStream_t stream) {
    const float* x      = (const float*)d_in[0];
    const float* ln3_w  = (const float*)d_in[1];
    const float* ln3_b  = (const float*)d_in[2];
    const float* qkv_w  = (const float*)d_in[3];
    const float* qkvdww = (const float*)d_in[4];
    const float* temp   = (const float*)d_in[5];
    const float* proj_w = (const float*)d_in[6];
    const float* ln4_w  = (const float*)d_in[7];
    const float* ln4_b  = (const float*)d_in[8];
    const float* pin_w  = (const float*)d_in[9];
    const float* ffn_dw = (const float*)d_in[10];
    const float* pout_w = (const float*)d_in[11];

    char* ws = (char*)d_ws;
    size_t off = 0;
    auto alloc = [&](size_t bytes) { void* p = ws + off; off = (off + bytes + 255) & ~(size_t)255; return p; };

    __bf16* wb_qkv  = (__bf16*)alloc((size_t)QKVC_ * C_ * 2);
    __bf16* wb_proj = (__bf16*)alloc((size_t)C_ * C_ * 2);
    __bf16* wb_pin  = (__bf16*)alloc((size_t)PINC_ * C_ * 2);
    __bf16* wb_pout = (__bf16*)alloc((size_t)C_ * HID_ * 2);
    __bf16* ln_t    = (__bf16*)alloc((size_t)NLOC_ * C_ * 2);
    float*  qkv     = (float*)alloc((size_t)QKVC_ * NLOC_ * 4);  // reused: attn-out bf16, pin bf16
    float*  qkvd    = (float*)alloc((size_t)QKVC_ * NLOC_ * 4);  // reused: gated y bf16
    float*  scaleQK = (float*)alloc((size_t)B_ * 384 * 4);
    float*  attnS   = (float*)alloc((size_t)24 * 1024 * 4);
    float*  x1      = (float*)alloc((size_t)C_ * NLOC_ * 4);
    (void)ws_size; (void)in_sizes; (void)n_in; (void)out_size;

    // weight conversion
    cvt_bf16_kernel<<<(QKVC_ * C_ + 255) / 256, 256, 0, stream>>>(qkv_w, wb_qkv, QKVC_ * C_);
    cvt_bf16_kernel<<<(C_ * C_ + 255) / 256, 256, 0, stream>>>(proj_w, wb_proj, C_ * C_);
    cvt_bf16_kernel<<<(PINC_ * C_ + 255) / 256, 256, 0, stream>>>(pin_w, wb_pin, PINC_ * C_);
    cvt_bf16_kernel<<<(C_ * HID_ + 255) / 256, 256, 0, stream>>>(pout_w, wb_pout, C_ * HID_);

    // --- attention branch ---
    ln_kernel<<<NLOC_ / 128, 128, 0, stream>>>(x, ln3_w, ln3_b, ln_t);
    gemm_kernel<<<dim3(NLOC_ / 256, QKVC_ / 64), 256, 0, stream>>>(
        ln_t, wb_qkv, C_, QKVC_, nullptr, qkv, nullptr);
    dwconv_kernel<<<(B_ * QKVC_ * HW_) / 256, 256, 0, stream>>>(qkv, qkvdww, qkvd, QKVC_);
    rownorm_kernel<<<B_ * 384, 256, 0, stream>>>(qkvd, scaleQK);
    attn_kernel<<<24, 256, 0, stream>>>(qkvd, scaleQK, temp, attnS);
    __bf16* attn_out = (__bf16*)qkv;                               // reuse
    attnv_kernel<<<24 * 64, 256, 0, stream>>>(qkvd, attnS, attn_out);
    gemm_kernel<<<dim3(NLOC_ / 256, C_ / 64), 256, 0, stream>>>(
        attn_out, wb_proj, C_, C_, x, x1, nullptr);

    // --- GDFN branch ---
    ln_kernel<<<NLOC_ / 128, 128, 0, stream>>>(x1, ln4_w, ln4_b, ln_t);
    __bf16* pin_out = (__bf16*)qkv;                                // reuse (bf16 NCHW, 768ch)
    gemm_kernel<<<dim3(NLOC_ / 256, PINC_ / 64), 256, 0, stream>>>(
        ln_t, wb_pin, C_, PINC_, nullptr, nullptr, pin_out);
    __bf16* y_t = (__bf16*)qkvd;                                   // reuse ([n][384] bf16)
    dwffn_gelu_kernel<<<(B_ * HID_ * HW_) / 256, 256, 0, stream>>>(pin_out, ffn_dw, y_t);
    gemm_kernel<<<dim3(NLOC_ / 256, C_ / 64), 256, 0, stream>>>(
        y_t, wb_pout, HID_, C_, x1, (float*)d_out, nullptr);
}
